// GraphSage_15530601743026
// MI455X (gfx1250) — matrix operand order repaired and checked by reading, not compiled, not run on hardware
//
#include <hip/hip_runtime.h>
#include <hip/hip_bf16.h>
#include <stdint.h>

// ---------------------------------------------------------------------------
// GraphSAGE 2-layer forward for MI455X (gfx1250, wave32).
//   layer1: h   = relu( mean_agg(x) @ w1l^T + b1 + x @ w1r^T )   [N,256]
//   layer2: out = log_softmax( mean_agg(h) @ w2l^T + b2 + h @ w2r^T )  [N,40]
// Data-movement in bf16, accumulation in f32 (WMMA f32_16x16x32_bf16).
// ---------------------------------------------------------------------------

#define NN    100000          // nodes
#define NE    1600000         // edges
#define DF    128             // input feature dim
#define DH    256             // hidden dim
#define NC    40              // classes
#define NCP   48              // classes padded to 3 x 16 tiles

typedef __attribute__((ext_vector_type(16))) __bf16 v16bf;
typedef __attribute__((ext_vector_type(8)))  float  v8f;

union Frag16 { uint4 q[2]; v16bf v; };
union Pack8  { uint4 q;    __bf16 h[8]; };
union Pack4  { uint2 q;    __bf16 h[4]; };

// ---------------- utility kernels ------------------------------------------

__global__ void k_zero_f4(float4* __restrict__ p, int n4) {
    int i = blockIdx.x * blockDim.x + threadIdx.x;
    if (i < n4) p[i] = make_float4(0.f, 0.f, 0.f, 0.f);
}

// f32 -> bf16, 8 elements per thread (n must be a multiple of 8)
__global__ void k_cvt_bf16_v8(const float* __restrict__ in,
                              __bf16* __restrict__ out, int n8) {
    int i = blockIdx.x * blockDim.x + threadIdx.x;
    if (i >= n8) return;
    const float4* p = (const float4*)in + (size_t)i * 2;
    float4 a = p[0], b = p[1];
    Pack8 U;
    U.h[0] = (__bf16)a.x; U.h[1] = (__bf16)a.y;
    U.h[2] = (__bf16)a.z; U.h[3] = (__bf16)a.w;
    U.h[4] = (__bf16)b.x; U.h[5] = (__bf16)b.y;
    U.h[6] = (__bf16)b.z; U.h[7] = (__bf16)b.w;
    ((uint4*)out)[i] = U.q;
}

// pad [NC,DH] f32 weight into [NCP,DH] bf16 (zero rows NC..NCP-1)
__global__ void k_pad_w2(const float* __restrict__ w, __bf16* __restrict__ out) {
    int i = blockIdx.x * blockDim.x + threadIdx.x;
    if (i >= NCP * DH) return;
    int row = i >> 8, col = i & 255;
    float v = (row < NC) ? w[row * DH + col] : 0.f;
    out[i] = (__bf16)v;
}

__global__ void k_pad_b2(const float* __restrict__ b, float* __restrict__ out) {
    int i = threadIdx.x;
    if (i < NCP) out[i] = (i < NC) ? b[i] : 0.f;
}

// agg[row, :] / max(cnt[row],1) -> bf16 ; 4 elems/thread ; rshift = log2(D)-2
__global__ void k_div_cvt(const float* __restrict__ agg, const float* __restrict__ cnt,
                          __bf16* __restrict__ out, int n4, int rshift) {
    int i = blockIdx.x * blockDim.x + threadIdx.x;
    if (i >= n4) return;
    int row = i >> rshift;
    float c = cnt[row];
    float inv = 1.f / (c > 1.f ? c : 1.f);
    float4 v = ((const float4*)agg)[i];
    Pack4 U;
    U.h[0] = (__bf16)(v.x * inv); U.h[1] = (__bf16)(v.y * inv);
    U.h[2] = (__bf16)(v.z * inv); U.h[3] = (__bf16)(v.w * inv);
    ((uint2*)out)[i] = U.q;
}

// ---------------- edge aggregation (wave per edge) --------------------------

// D = 128: lane covers 4 bf16 (8 B gather), 4 f32 atomics
__global__ void k_aggregate128(const __bf16* __restrict__ xb,
                               const int* __restrict__ src, const int* __restrict__ dst,
                               float* __restrict__ agg, float* __restrict__ cnt) {
    int wid  = (blockIdx.x * blockDim.x + threadIdx.x) >> 5;
    int lane = threadIdx.x & 31;
    if (wid >= NE) return;
    int s = src[wid], d = dst[wid];
    Pack4 U; U.q = *(const uint2*)(xb + (size_t)s * DF + lane * 4);
    float* ap = agg + (size_t)d * DF + lane * 4;
#pragma unroll
    for (int i = 0; i < 4; ++i) atomicAdd(ap + i, (float)U.h[i]);
    if (lane == 0) atomicAdd(cnt + d, 1.0f);
}

// D = 256: lane covers 8 bf16 (16 B gather), 8 f32 atomics
__global__ void k_aggregate256(const __bf16* __restrict__ hb,
                               const int* __restrict__ src, const int* __restrict__ dst,
                               float* __restrict__ agg) {
    int wid  = (blockIdx.x * blockDim.x + threadIdx.x) >> 5;
    int lane = threadIdx.x & 31;
    if (wid >= NE) return;
    int s = src[wid], d = dst[wid];
    Pack8 U; U.q = *(const uint4*)(hb + (size_t)s * DH + lane * 8);
    float* ap = agg + (size_t)d * DH + lane * 8;
#pragma unroll
    for (int i = 0; i < 8; ++i) atomicAdd(ap + i, (float)U.h[i]);
}

// ---------------- WMMA fragment loads ---------------------------------------

// A 16x32 bf16 (ISA 7.12.2): lane m -> row m, K = {kh*8..+7} U {16+kh*8..+7}
__device__ __forceinline__ v16bf load_a(const __bf16* base, int K, int row,
                                        int kh, int kb) {
    Frag16 f;
    const __bf16* p = base + (size_t)row * K + kb + kh * 8;
    f.q[0] = *(const uint4*)p;
    f.q[1] = *(const uint4*)(p + 16);
    return f.v;
}
// B 32x16 bf16: lane n -> col n, K = {kh*8..+7} U {16+kh*8..+7}
// B[k,n] = W[n,k] (row-major weight row) -> straight contiguous load
__device__ __forceinline__ v16bf load_b(const __bf16* base, int K, int col,
                                        int kh, int kb) {
    Frag16 f;
    const __bf16* p = base + (size_t)col * K + kb + kh * 8;
    f.q[0] = *(const uint4*)p;
    f.q[1] = *(const uint4*)(p + 16);
    return f.v;
}

// ---------------- GEMM 1: [NN,256] = relu(agg1b@W1l^T + b1 + xb@W1r^T) ------

__global__ void k_gemm1(const __bf16* __restrict__ A1,   // agg1b [NN,128]
                        const __bf16* __restrict__ W1,   // w1l   [256,128]
                        const __bf16* __restrict__ A2,   // xb    [NN,128]
                        const __bf16* __restrict__ W2,   // w1r   [256,128]
                        const float*  __restrict__ bias, // b1    [256]
                        __bf16* __restrict__ out) {      // h_b   [NN,256]
    int lane = threadIdx.x & 31;
    int tile = (blockIdx.x * blockDim.x + threadIdx.x) >> 5;
    int tM = tile >> 4;        // 16 N-tiles (256/16)
    int tN = tile & 15;
    if (tM >= NN / 16) return;
    int r  = lane & 15, kh = lane >> 4;
    int row = tM * 16 + r;
    int col = tN * 16 + r;
    v8f c = {};
#pragma unroll
    for (int kb = 0; kb < DF; kb += 32) {
        v16bf a = load_a(A1, DF, row, kh, kb);
        v16bf b = load_b(W1, DF, col, kh, kb);
        c = __builtin_amdgcn_wmma_f32_16x16x32_bf16(false, a, false, b,
                                                    (short)0, c, false, false);
    }
#pragma unroll
    for (int kb = 0; kb < DF; kb += 32) {
        v16bf a = load_a(A2, DF, row, kh, kb);
        v16bf b = load_b(W2, DF, col, kh, kb);
        c = __builtin_amdgcn_wmma_f32_16x16x32_bf16(false, a, false, b,
                                                    (short)0, c, false, false);
    }
    // C/D layout: vgpr i -> M = i + (lane>>4)*8, N = lane&15
    int n = tN * 16 + (lane & 15);
    float bv = bias[n];
    int mbase = tM * 16 + (lane >> 4) * 8;
#pragma unroll
    for (int i = 0; i < 8; ++i) {
        float v = c[i] + bv;
        v = v > 0.f ? v : 0.f;
        out[(size_t)(mbase + i) * DH + n] = (__bf16)v;
    }
}

// ---------------- GEMM 2: pre[NN,48] = agg2b@W2l^T + b2 + hb@W2r^T ----------

__global__ void k_gemm2(const __bf16* __restrict__ A1,   // agg2b [NN,256]
                        const __bf16* __restrict__ W1,   // w2l pad [48,256]
                        const __bf16* __restrict__ A2,   // h_b   [NN,256]
                        const __bf16* __restrict__ W2,   // w2r pad [48,256]
                        const float*  __restrict__ bias, // b2 pad [48]
                        float* __restrict__ out) {       // pre  [NN,48]
    int lane = threadIdx.x & 31;
    int tile = (blockIdx.x * blockDim.x + threadIdx.x) >> 5;
    if (tile >= (NN / 16) * 3) return;
    int tM = tile / 3;
    int tN = tile % 3;
    int r  = lane & 15, kh = lane >> 4;
    int row = tM * 16 + r;
    int col = tN * 16 + r;
    v8f c = {};
#pragma unroll
    for (int kb = 0; kb < DH; kb += 32) {
        v16bf a = load_a(A1, DH, row, kh, kb);
        v16bf b = load_b(W1, DH, col, kh, kb);
        c = __builtin_amdgcn_wmma_f32_16x16x32_bf16(false, a, false, b,
                                                    (short)0, c, false, false);
    }
#pragma unroll
    for (int kb = 0; kb < DH; kb += 32) {
        v16bf a = load_a(A2, DH, row, kh, kb);
        v16bf b = load_b(W2, DH, col, kh, kb);
        c = __builtin_amdgcn_wmma_f32_16x16x32_bf16(false, a, false, b,
                                                    (short)0, c, false, false);
    }
    int n = tN * 16 + (lane & 15);
    float bv = bias[n];
    int mbase = tM * 16 + (lane >> 4) * 8;
#pragma unroll
    for (int i = 0; i < 8; ++i)
        out[(size_t)(mbase + i) * NCP + n] = c[i] + bv;
}

// ---------------- log_softmax over 40 classes (wave per row) ----------------

__global__ void k_log_softmax(const float* __restrict__ pre,
                              float* __restrict__ out) {
    int wid  = (blockIdx.x * blockDim.x + threadIdx.x) >> 5;
    int lane = threadIdx.x & 31;
    if (wid >= NN) return;
    const float* rp = pre + (size_t)wid * NCP;
    float v0 = rp[lane];                                    // cols 0..31
    float v1 = (lane < 8) ? rp[lane + 32] : -__builtin_inff(); // cols 32..39
    float m = fmaxf(v0, v1);
#pragma unroll
    for (int off = 16; off; off >>= 1) m = fmaxf(m, __shfl_xor(m, off, 32));
    float s = __expf(v0 - m) + ((lane < 8) ? __expf(v1 - m) : 0.f);
#pragma unroll
    for (int off = 16; off; off >>= 1) s += __shfl_xor(s, off, 32);
    float lse = __logf(s);
    float* op = out + (size_t)wid * NC;
    op[lane] = v0 - m - lse;
    if (lane < 8) op[lane + 32] = v1 - m - lse;
}

// ---------------------------------------------------------------------------

extern "C" void kernel_launch(void* const* d_in, const int* in_sizes, int n_in,
                              void* d_out, int out_size, void* d_ws, size_t ws_size,
                              hipStream_t stream) {
    const float* x   = (const float*)d_in[0];
    const int*   ei  = (const int*)  d_in[1];
    const float* w1l = (const float*)d_in[2];
    const float* b1  = (const float*)d_in[3];
    const float* w1r = (const float*)d_in[4];
    const float* w2l = (const float*)d_in[5];
    const float* b2  = (const float*)d_in[6];
    const float* w2r = (const float*)d_in[7];
    const int* srcI = ei;
    const int* dstI = ei + NE;

    // -------- workspace layout (aliased: phase-1 buffers die before agg2) ---
    char* ws = (char*)d_ws;
    size_t o = 0;
    auto take = [&](size_t bytes) { size_t r = o; o += (bytes + 255) & ~(size_t)255; return r; };
    size_t o_cnt  = take((size_t)NN * 4);             // degree counts (both layers)
    size_t o_R    = take((size_t)NN * DH * 4);        // shared 102.4MB region
    size_t o_xb    = o_R;                             //   xb    [NN,128] bf16
    size_t o_agg1  = o_R + (size_t)NN * DF * 2;       //   agg1  [NN,128] f32
    size_t o_agg1b = o_agg1 + (size_t)NN * DF * 4;    //   agg1b [NN,128] bf16
    size_t o_agg2  = o_R;                             //   agg2  [NN,256] f32 (reuse)
    size_t o_hb   = take((size_t)NN * DH * 2);        // h bf16 [NN,256]
    size_t o_a2b  = take((size_t)NN * DH * 2);        // agg2b  [NN,256] bf16
    size_t o_pre  = take((size_t)NN * NCP * 4);       // pre-softmax [NN,48] f32
    size_t o_w1lb = take((size_t)DH * DF * 2);
    size_t o_w1rb = take((size_t)DH * DF * 2);
    size_t o_w2lb = take((size_t)NCP * DH * 2);
    size_t o_w2rb = take((size_t)NCP * DH * 2);
    size_t o_b2p  = take((size_t)NCP * 4);
    (void)ws_size; (void)in_sizes; (void)n_in; (void)out_size;

    float*  cnt   = (float*) (ws + o_cnt);
    __bf16* xb    = (__bf16*)(ws + o_xb);
    float*  agg1  = (float*) (ws + o_agg1);
    __bf16* agg1b = (__bf16*)(ws + o_agg1b);
    float*  agg2  = (float*) (ws + o_agg2);
    __bf16* hb    = (__bf16*)(ws + o_hb);
    __bf16* a2b   = (__bf16*)(ws + o_a2b);
    float*  pre   = (float*) (ws + o_pre);
    __bf16* w1lb  = (__bf16*)(ws + o_w1lb);
    __bf16* w1rb  = (__bf16*)(ws + o_w1rb);
    __bf16* w2lb  = (__bf16*)(ws + o_w2lb);
    __bf16* w2rb  = (__bf16*)(ws + o_w2rb);
    float*  b2p   = (float*) (ws + o_b2p);

    const int T = 256;

    // 1) zero layer-1 accumulators + degree counts
    k_zero_f4<<<(NN * DF / 4 + T - 1) / T, T, 0, stream>>>((float4*)agg1, NN * DF / 4);
    k_zero_f4<<<(NN / 4 + T - 1) / T, T, 0, stream>>>((float4*)cnt, NN / 4);

    // 2) convert features + weights to bf16 (pad class weights to 48 rows)
    k_cvt_bf16_v8<<<(NN * DF / 8 + T - 1) / T, T, 0, stream>>>(x, xb, NN * DF / 8);
    k_cvt_bf16_v8<<<(DH * DF / 8 + T - 1) / T, T, 0, stream>>>(w1l, w1lb, DH * DF / 8);
    k_cvt_bf16_v8<<<(DH * DF / 8 + T - 1) / T, T, 0, stream>>>(w1r, w1rb, DH * DF / 8);
    k_pad_w2<<<(NCP * DH + T - 1) / T, T, 0, stream>>>(w2l, w2lb);
    k_pad_w2<<<(NCP * DH + T - 1) / T, T, 0, stream>>>(w2r, w2rb);
    k_pad_b2<<<1, 64, 0, stream>>>(b2, b2p);

    // 3) layer-1 mean aggregation (wave per edge) + normalize to bf16
    k_aggregate128<<<NE / 8, T, 0, stream>>>(xb, srcI, dstI, agg1, cnt);
    k_div_cvt<<<(NN * DF / 4 + T - 1) / T, T, 0, stream>>>(agg1, cnt, agg1b,
                                                          NN * DF / 4, 5);

    // 4) layer-1 WMMA GEMM + bias + ReLU -> h (bf16)
    k_gemm1<<<(NN / 16) * 16 / 8, T, 0, stream>>>(agg1b, w1lb, xb, w1rb, b1, hb);

    // 5) layer-2 aggregation (agg2 reuses the phase-1 region) + normalize
    k_zero_f4<<<(NN * DH / 4 + T - 1) / T, T, 0, stream>>>((float4*)agg2, NN * DH / 4);
    k_aggregate256<<<NE / 8, T, 0, stream>>>(hb, srcI, dstI, agg2);
    k_div_cvt<<<(NN * DH / 4 + T - 1) / T, T, 0, stream>>>(agg2, cnt, a2b,
                                                          NN * DH / 4, 6);

    // 6) layer-2 WMMA GEMM (padded to 48 cols) + bias
    k_gemm2<<<((NN / 16) * 3 + 7) / 8, T, 0, stream>>>(a2b, w2lb, hb, w2rb, b2p, pre);

    // 7) log_softmax over 40 classes
    k_log_softmax<<<NN / 8, T, 0, stream>>>(pre, (float*)d_out);
}
